// GPT_50156628083201
// MI455X (gfx1250) — compile-verified, matrix operand order checked by reading, not run on hardware
//
#include <hip/hip_runtime.h>
#include <stdint.h>

#define BB 4
#define TT 1024
#define CC 768
#define HH 12
#define DD 64
#define VV 50257
#define LL 12
#define MM (BB * TT)

typedef __attribute__((ext_vector_type(16))) __bf16 v16bf;
typedef __attribute__((ext_vector_type(8)))  float  v8f;
typedef unsigned int u32x4 __attribute__((ext_vector_type(4)));
typedef int          i32x4 __attribute__((ext_vector_type(4)));
typedef int          i32x8 __attribute__((ext_vector_type(8)));

struct alignas(16) U128 { unsigned int u[4]; };
union FragBF { U128 q[2]; unsigned short s[16]; v16bf v; };

#if defined(__has_builtin)
#  if __has_builtin(__builtin_amdgcn_tensor_load_to_lds)
#    define HAVE_TDM 1
#  endif
#endif
#ifndef HAVE_TDM
#  define HAVE_TDM 0
#endif

__device__ inline unsigned short f2bf(float f) {
    unsigned int u = __float_as_uint(f);
    u = (u + 0x7FFFu + ((u >> 16) & 1u)) >> 16;   // round-to-nearest-even
    return (unsigned short)u;
}

#if HAVE_TDM
// Issue one TDM DMA: 128x32-ushort tile of a [rows][K] row-major bf16 tensor
// into LDS with a 4-DWORD pad every 16 DWORDs (=> PK=40 ushort LDS stride).
__device__ inline void tdm_load_tile(const unsigned short* gptr, unsigned lds_byte_off,
                                     int kdim, int rows_remaining) {
    const unsigned long long ga = (unsigned long long)(uintptr_t)gptr;
    u32x4 g0;
    g0[0] = 1u;                                            // count=1, user descriptor
    g0[1] = lds_byte_off;                                  // D#.lds_addr
    g0[2] = (unsigned)(ga & 0xffffffffu);                  // global_addr[31:0]
    g0[3] = (unsigned)((ga >> 32) & 0x1ffffffu) | (2u << 30);  // [56:32] | type=2
    i32x8 g1;
    //        data_size=2B      pad_en     pad_interval=16DW pad_amount=4DW
    g1[0] = (int)((1u << 16) | (1u << 20) | (3u << 22) | (3u << 25));
    g1[1] = (int)(((unsigned)kdim & 0xffffu) << 16);               // tensor_dim0 lo16
    g1[2] = (int)(((unsigned)kdim >> 16) |
                  (((unsigned)rows_remaining & 0xffffu) << 16));   // dim0 hi | dim1 lo16
    g1[3] = (int)(((unsigned)rows_remaining >> 16) | (32u << 16)); // dim1 hi | tile_dim0=32
    g1[4] = (int)(128u);                                           // tile_dim1=128, tile_dim2=0
    g1[5] = (int)(unsigned)kdim;                                   // tensor_dim0_stride lo32
    g1[6] = 0;                                                     // stride0 hi | stride1 lo
    g1[7] = 0;
    const i32x4 z4 = {0, 0, 0, 0};
#if __clang_major__ <= 22
    __builtin_amdgcn_tensor_load_to_lds(g0, g1, z4, z4, 0);
#else
    const i32x8 z8 = {0, 0, 0, 0, 0, 0, 0, 0};
    __builtin_amdgcn_tensor_load_to_lds(g0, g1, z4, z4, z8, 0);
#endif
}
#endif

// ---------------------------------------------------------------- conversions
__global__ __launch_bounds__(256)
void cvt_kernel(const float* __restrict__ in, unsigned short* __restrict__ out, size_t n) {
    for (size_t i = (size_t)blockIdx.x * blockDim.x + threadIdx.x; i < n;
         i += (size_t)gridDim.x * blockDim.x)
        out[i] = f2bf(in[i]);
}

// transpose + convert: in [L][R][Cn] f32  ->  out [L][Cn][R] bf16  (R, Cn multiples of 32)
__global__ __launch_bounds__(256)
void cvt_t_kernel(const float* __restrict__ in, unsigned short* __restrict__ out, int R, int Cn) {
    __shared__ float tile[32][33];
    const int l = blockIdx.z;
    const float* src = in + (size_t)l * R * Cn;
    unsigned short* dst = out + (size_t)l * R * Cn;
    const int r0 = blockIdx.y * 32, c0 = blockIdx.x * 32;
    const int tx = threadIdx.x & 31, ty = threadIdx.x >> 5;
    for (int r = ty; r < 32; r += 8)
        tile[r][tx] = src[(size_t)(r0 + r) * Cn + c0 + tx];
    __syncthreads();
    for (int r = ty; r < 32; r += 8)
        dst[(size_t)(c0 + r) * R + r0 + tx] = f2bf(tile[tx][r]);
}

// ---------------------------------------------------------------- embedding
__global__ __launch_bounds__(256)
void embed_kernel(const int* __restrict__ idx, const float* __restrict__ wte,
                  const float* __restrict__ wpe, float* __restrict__ x) {
    const int t = blockIdx.x;                 // 0 .. MM-1  (== b*T + pos)
    const int pos = t & (TT - 1);
    const size_t id = (size_t)idx[t];
    for (int c = threadIdx.x; c < CC; c += 256)
        x[(size_t)t * CC + c] = wte[id * CC + c] + wpe[(size_t)pos * CC + c];
}

// ---------------------------------------------------------------- layernorm (wave per token)
__global__ __launch_bounds__(256)
void ln_kernel(const float* __restrict__ x, const float* __restrict__ sc,
               const float* __restrict__ bi, unsigned short* __restrict__ h) {
    const int lane = threadIdx.x & 31;
    const int tok  = blockIdx.x * 8 + (threadIdx.x >> 5);
    const float* xr = x + (size_t)tok * CC;
    float sum = 0.f, sq = 0.f;
    for (int i = lane; i < CC; i += 32) { float v = xr[i]; sum += v; sq += v * v; }
    #pragma unroll
    for (int off = 1; off < 32; off <<= 1) {
        sum += __shfl_xor(sum, off, 32);
        sq  += __shfl_xor(sq,  off, 32);
    }
    const float mean = sum * (1.f / CC);
    const float var  = sq * (1.f / CC) - mean * mean;
    const float inv  = rsqrtf(var + 1e-5f);
    for (int i = lane; i < CC; i += 32)
        h[(size_t)tok * CC + i] = f2bf((xr[i] - mean) * inv * sc[i] + bi[i]);
}

// ---------------------------------------------------------------- bf16 WMMA GEMM
// A: [M][K] bf16 row-major.  Bt: [N][K] bf16 (K-contiguous per output column).
// OP: 0 = f32 store, 1 = bf16 store, 2 = GELU + bf16 store, 3 = f32 residual add
template <int OP>
__global__ __launch_bounds__(256)
void gemm_bf16(const unsigned short* __restrict__ A, const unsigned short* __restrict__ Bt,
               const float* __restrict__ bias, void* __restrict__ OutP,
               int M, int N, int K) {
    constexpr int BK = 32, PK = 40;           // PK: padded LDS K-stride (TDM pad fields match)
    __shared__ alignas(16) unsigned short As[2][128 * PK];
    __shared__ alignas(16) unsigned short Bs[2][128 * PK];

    const int tid  = threadIdx.x;
    const int lane = tid & 31;
    const int wave = tid >> 5;
    const int hf   = lane >> 4;               // wave half (C-layout row group)
    const int l15  = lane & 15;
    const int wm   = (wave & 1) * 64;         // 2 waves along M
    const int wn   = (wave >> 1) * 32;        // 4 waves along N
    const int m0   = blockIdx.y * 128;
    const int n0   = blockIdx.x * 128;

    const v8f vz = {0.f,0.f,0.f,0.f,0.f,0.f,0.f,0.f};
    v8f acc[4][2];
    #pragma unroll
    for (int i = 0; i < 4; ++i)
        #pragma unroll
        for (int j = 0; j < 2; ++j) acc[i][j] = vz;

#if HAVE_TDM
    // --- TDM double-buffered pipeline: wave 0 drives the Tensor Data Mover ---
    const bool issuer = (wave == 0);
    const int  nrem   = N - n0;               // TDM zero-fills rows beyond tensor_dim1
    if (issuer) {
        tdm_load_tile(A  + (size_t)m0 * K, (unsigned)(uintptr_t)&As[0][0], K, M - m0);
        tdm_load_tile(Bt + (size_t)n0 * K, (unsigned)(uintptr_t)&Bs[0][0], K, nrem);
    }
    for (int kb = 0; kb < K; kb += BK) {
        const int cur = (kb >> 5) & 1;
        if (issuer) __builtin_amdgcn_s_wait_tensorcnt(0);
        __syncthreads();                      // tiles[cur] ready; prior readers of [cur^1] done
        if (issuer && kb + BK < K) {
            tdm_load_tile(A  + (size_t)m0 * K + kb + BK,
                          (unsigned)(uintptr_t)&As[cur ^ 1][0], K, M - m0);
            tdm_load_tile(Bt + (size_t)n0 * K + kb + BK,
                          (unsigned)(uintptr_t)&Bs[cur ^ 1][0], K, nrem);
        }
        const unsigned short* as = As[cur];
        const unsigned short* bs = Bs[cur];
#else
    // --- fallback: per-thread vector staging ---
    const int lr = tid >> 1;                  // 0..127 staging row
    const int lc = (tid & 1) * 16;            // 0 or 16
    for (int kb = 0; kb < K; kb += BK) {
        {
            const U128* g = (const U128*)(A + (size_t)(m0 + lr) * K + kb + lc);
            *(U128*)(&As[0][0] + lr * PK + lc)     = g[0];
            *(U128*)(&As[0][0] + lr * PK + lc + 8) = g[1];
        }
        {
            U128 z0 = {{0,0,0,0}}, z1 = {{0,0,0,0}};
            if (n0 + lr < N) {
                const U128* g = (const U128*)(Bt + (size_t)(n0 + lr) * K + kb + lc);
                z0 = g[0]; z1 = g[1];
            }
            *(U128*)(&Bs[0][0] + lr * PK + lc)     = z0;
            *(U128*)(&Bs[0][0] + lr * PK + lc + 8) = z1;
        }
        __syncthreads();
        const unsigned short* as = &As[0][0];
        const unsigned short* bs = &Bs[0][0];
#endif
        FragBF bf[2];
        #pragma unroll
        for (int j = 0; j < 2; ++j) {
            const int nb = (wn + j * 16 + l15) * PK;
            bf[j].q[0] = *(const U128*)(bs + nb + hf * 8);
            bf[j].q[1] = *(const U128*)(bs + nb + 16 + hf * 8);
        }
        #pragma unroll
        for (int i = 0; i < 4; ++i) {
            FragBF af;
            const int ab = (wm + i * 16 + l15) * PK;
            af.q[0] = *(const U128*)(as + ab + hf * 8);
            af.q[1] = *(const U128*)(as + ab + 16 + hf * 8);
            #pragma unroll
            for (int j = 0; j < 2; ++j)
                acc[i][j] = __builtin_amdgcn_wmma_f32_16x16x32_bf16(
                    false, af.v, false, bf[j].v, (short)0, acc[i][j], false, false);
        }
#if !HAVE_TDM
        __syncthreads();
#endif
    }

    // epilogue (C layout: lane half -> rows 0-7 / 8-15, lanes hold columns)
    #pragma unroll
    for (int i = 0; i < 4; ++i) {
        #pragma unroll
        for (int j = 0; j < 2; ++j) {
            #pragma unroll
            for (int r = 0; r < 8; ++r) {
                const int row = m0 + wm + i * 16 + hf * 8 + r;
                const int col = n0 + wn + j * 16 + l15;
                if (col < N) {
                    float v = acc[i][j][r] + (bias ? bias[col] : 0.f);
                    const size_t o = (size_t)row * N + col;
                    if (OP == 0) {
                        ((float*)OutP)[o] = v;
                    } else if (OP == 1) {
                        ((unsigned short*)OutP)[o] = f2bf(v);
                    } else if (OP == 2) {
                        float g = 0.5f * v * (1.f + tanhf(0.7978845608f * (v + 0.044715f * v * v * v)));
                        ((unsigned short*)OutP)[o] = f2bf(g);
                    } else {
                        ((float*)OutP)[o] += v;    // residual accumulate
                    }
                }
            }
        }
    }
}

// ---------------------------------------------------------------- flash attention
// one wave handles 16 query rows for one (b, h); streams 16-key blocks
__global__ __launch_bounds__(32)
void attn_kernel(const unsigned short* __restrict__ qkv, unsigned short* __restrict__ o) {
    __shared__ alignas(16) unsigned short Pst[16 * 16];
    const int lane = threadIdx.x & 31;
    const int hf   = lane >> 4;
    const int l15  = lane & 15;
    const int q0   = blockIdx.x * 16;
    const int h    = blockIdx.y;
    const int b    = blockIdx.z;
    const size_t tok0 = (size_t)b * TT;
    const int C3 = 3 * CC;

    FragBF qf[2];
    {
        const unsigned short* qp = qkv + (tok0 + q0 + l15) * C3 + h * DD;
        #pragma unroll
        for (int c = 0; c < 2; ++c) {
            qf[c].q[0] = *(const U128*)(qp + c * 32 + hf * 8);
            qf[c].q[1] = *(const U128*)(qp + c * 32 + 16 + hf * 8);
        }
    }

    float mrow[8], lrow[8];
    #pragma unroll
    for (int r = 0; r < 8; ++r) { mrow[r] = -1e30f; lrow[r] = 0.f; }
    const v8f vz = {0.f,0.f,0.f,0.f,0.f,0.f,0.f,0.f};
    v8f oacc[4];
    #pragma unroll
    for (int nf = 0; nf < 4; ++nf) oacc[nf] = vz;

    const int nks = q0 / 16 + 1;                  // causal: keys up to q0+15
    for (int kt = 0; kt < nks; ++kt) {
        const int ks = kt * 16;
        FragBF kf[2];
        {
            const unsigned short* kp = qkv + (tok0 + ks + l15) * C3 + CC + h * DD;
            #pragma unroll
            for (int c = 0; c < 2; ++c) {
                kf[c].q[0] = *(const U128*)(kp + c * 32 + hf * 8);
                kf[c].q[1] = *(const U128*)(kp + c * 32 + 16 + hf * 8);
            }
        }
        v8f s = vz;
        s = __builtin_amdgcn_wmma_f32_16x16x32_bf16(false, qf[0].v, false, kf[0].v, (short)0, s, false, false);
        s = __builtin_amdgcn_wmma_f32_16x16x32_bf16(false, qf[1].v, false, kf[1].v, (short)0, s, false, false);

        const int colk = ks + l15;
        float pr[8], corr[8];
        #pragma unroll
        for (int r = 0; r < 8; ++r) {
            const int mg = q0 + hf * 8 + r;
            float sv = (colk <= mg) ? s[r] * 0.125f : -1e30f;   // 1/sqrt(64), causal mask
            float rm = sv;
            #pragma unroll
            for (int off = 1; off < 16; off <<= 1) rm = fmaxf(rm, __shfl_xor(rm, off, 32));
            const float mn = fmaxf(mrow[r], rm);
            const float p  = __expf(sv - mn);
            float ps = p;
            #pragma unroll
            for (int off = 1; off < 16; off <<= 1) ps += __shfl_xor(ps, off, 32);
            corr[r] = __expf(mrow[r] - mn);
            lrow[r] = lrow[r] * corr[r] + ps;
            mrow[r] = mn;
            pr[r]   = p;
        }
        #pragma unroll
        for (int nf = 0; nf < 4; ++nf)
            #pragma unroll
            for (int r = 0; r < 8; ++r) oacc[nf][r] *= corr[r];

        #pragma unroll
        for (int r = 0; r < 8; ++r) Pst[(hf * 8 + r) * 16 + l15] = f2bf(pr[r]);
        __syncthreads();
        FragBF pf;
        pf.q[0] = *(const U128*)(Pst + l15 * 16 + hf * 8);
        pf.q[1].u[0] = 0; pf.q[1].u[1] = 0; pf.q[1].u[2] = 0; pf.q[1].u[3] = 0;

        #pragma unroll
        for (int nf = 0; nf < 4; ++nf) {
            FragBF vf;
            const int d = nf * 16 + l15;
            #pragma unroll
            for (int jj = 0; jj < 8; ++jj) {
                const int key = ks + hf * 8 + jj;
                vf.s[jj] = qkv[(tok0 + key) * C3 + 2 * CC + h * DD + d];
            }
            #pragma unroll
            for (int jj = 8; jj < 16; ++jj) vf.s[jj] = 0;   // matches zero-padded A half
            oacc[nf] = __builtin_amdgcn_wmma_f32_16x16x32_bf16(
                false, pf.v, false, vf.v, (short)0, oacc[nf], false, false);
        }
        __syncthreads();
    }

    #pragma unroll
    for (int nf = 0; nf < 4; ++nf) {
        const int d = nf * 16 + l15;
        #pragma unroll
        for (int r = 0; r < 8; ++r) {
            const int m = q0 + hf * 8 + r;
            o[(tok0 + m) * CC + h * DD + d] = f2bf(oacc[nf][r] / lrow[r]);
        }
    }
}

// ---------------------------------------------------------------- loss
__global__ __launch_bounds__(256)
void loss_row_kernel(const float* __restrict__ logits, const int* __restrict__ tgt,
                     float* __restrict__ rowloss) {
    __shared__ float red[256];
    const int row = blockIdx.x;
    const int tid = threadIdx.x;
    const float* lr = logits + (size_t)row * VV;
    float mx = -1e30f;
    for (int i = tid; i < VV; i += 256) mx = fmaxf(mx, lr[i]);
    red[tid] = mx; __syncthreads();
    for (int s = 128; s > 0; s >>= 1) { if (tid < s) red[tid] = fmaxf(red[tid], red[tid + s]); __syncthreads(); }
    const float m = red[0]; __syncthreads();
    float se = 0.f;
    for (int i = tid; i < VV; i += 256) se += __expf(lr[i] - m);
    red[tid] = se; __syncthreads();
    for (int s = 128; s > 0; s >>= 1) { if (tid < s) red[tid] += red[tid + s]; __syncthreads(); }
    if (tid == 0) rowloss[row] = logf(red[0]) + m - lr[tgt[row]];
}

__global__ __launch_bounds__(256)
void loss_mean_kernel(const float* __restrict__ rowloss, float* __restrict__ out) {
    __shared__ float red[256];
    float s = 0.f;
    for (int i = threadIdx.x; i < MM; i += 256) s += rowloss[i];
    red[threadIdx.x] = s; __syncthreads();
    for (int k = 128; k > 0; k >>= 1) { if (threadIdx.x < k) red[threadIdx.x] += red[threadIdx.x + k]; __syncthreads(); }
    if (threadIdx.x == 0) out[(size_t)MM * VV] = red[0] / (float)MM;
}

// ---------------------------------------------------------------- launcher
extern "C" void kernel_launch(void* const* d_in, const int* in_sizes, int n_in,
                              void* d_out, int out_size, void* d_ws, size_t ws_size,
                              hipStream_t stream) {
    (void)in_sizes; (void)n_in; (void)out_size; (void)ws_size;
    const int*   idx   = (const int*)d_in[0];
    const int*   tgt   = (const int*)d_in[1];
    const float* wte   = (const float*)d_in[2];
    const float* wpe   = (const float*)d_in[3];
    const float* ln1_s = (const float*)d_in[4];
    const float* ln1_b = (const float*)d_in[5];
    const float* Wqkv  = (const float*)d_in[6];
    const float* bqkv  = (const float*)d_in[7];
    const float* Wproj = (const float*)d_in[8];
    const float* bproj = (const float*)d_in[9];
    const float* ln2_s = (const float*)d_in[10];
    const float* ln2_b = (const float*)d_in[11];
    const float* Wfc   = (const float*)d_in[12];
    const float* bfc   = (const float*)d_in[13];
    const float* Wcp   = (const float*)d_in[14];
    const float* bcp   = (const float*)d_in[15];
    const float* lnf_s = (const float*)d_in[16];
    const float* lnf_b = (const float*)d_in[17];
    float* out = (float*)d_out;

    char* ws = (char*)d_ws;
    size_t off = 0;
    auto take = [&](size_t bytes) -> char* {
        char* p = ws + off; off = (off + bytes + 255) & ~(size_t)255; return p;
    };
    unsigned short* wteBF   = (unsigned short*)take((size_t)VV * CC * 2);        // [V][C] (already [N][K])
    unsigned short* WqkvT   = (unsigned short*)take((size_t)LL * CC * 3 * CC * 2); // [3C][C]
    unsigned short* WprojT  = (unsigned short*)take((size_t)LL * CC * CC * 2);     // [C][C]
    unsigned short* WfcT    = (unsigned short*)take((size_t)LL * CC * 4 * CC * 2); // [4C][C]
    unsigned short* WcpT    = (unsigned short*)take((size_t)LL * 4 * CC * CC * 2); // [C][4C]
    float*          xbuf    = (float*)take((size_t)MM * CC * 4);
    unsigned short* hbuf    = (unsigned short*)take((size_t)MM * CC * 2);
    unsigned short* qkvbuf  = (unsigned short*)take((size_t)MM * 3 * CC * 2);
    unsigned short* attno   = (unsigned short*)take((size_t)MM * CC * 2);
    unsigned short* fcbuf   = (unsigned short*)take((size_t)MM * 4 * CC * 2);
    float*          rowloss = (float*)take((size_t)MM * 4);

    // one-time weight prep: bf16 convert (+ transpose to [N][K] for the TDM path)
    cvt_kernel<<<2048, 256, 0, stream>>>(wte, wteBF, (size_t)VV * CC);
    cvt_t_kernel<<<dim3(3 * CC / 32, CC / 32, LL),     256, 0, stream>>>(Wqkv,  WqkvT,  CC,     3 * CC);
    cvt_t_kernel<<<dim3(CC / 32,     CC / 32, LL),     256, 0, stream>>>(Wproj, WprojT, CC,     CC);
    cvt_t_kernel<<<dim3(4 * CC / 32, CC / 32, LL),     256, 0, stream>>>(Wfc,   WfcT,   CC,     4 * CC);
    cvt_t_kernel<<<dim3(CC / 32,     4 * CC / 32, LL), 256, 0, stream>>>(Wcp,   WcpT,   4 * CC, CC);

    embed_kernel<<<MM, 256, 0, stream>>>(idx, wte, wpe, xbuf);

    const dim3 gq((3 * CC + 127) / 128, MM / 128);
    const dim3 gp((CC + 127) / 128,     MM / 128);
    const dim3 gf((4 * CC + 127) / 128, MM / 128);
    const dim3 gl((VV + 127) / 128,     MM / 128);
    const dim3 ga(TT / 16, HH, BB);

    for (int l = 0; l < LL; ++l) {
        ln_kernel<<<MM / 8, 256, 0, stream>>>(xbuf, ln1_s + l * CC, ln1_b + l * CC, hbuf);
        gemm_bf16<1><<<gq, 256, 0, stream>>>(hbuf, WqkvT + (size_t)l * CC * 3 * CC,
                                             bqkv + (size_t)l * 3 * CC, qkvbuf, MM, 3 * CC, CC);
        attn_kernel<<<ga, 32, 0, stream>>>(qkvbuf, attno);
        gemm_bf16<3><<<gp, 256, 0, stream>>>(attno, WprojT + (size_t)l * CC * CC,
                                             bproj + (size_t)l * CC, xbuf, MM, CC, CC);
        ln_kernel<<<MM / 8, 256, 0, stream>>>(xbuf, ln2_s + l * CC, ln2_b + l * CC, hbuf);
        gemm_bf16<2><<<gf, 256, 0, stream>>>(hbuf, WfcT + (size_t)l * CC * 4 * CC,
                                             bfc + (size_t)l * 4 * CC, fcbuf, MM, 4 * CC, CC);
        gemm_bf16<3><<<gp, 256, 0, stream>>>(fcbuf, WcpT + (size_t)l * 4 * CC * CC,
                                             bcp + (size_t)l * CC, xbuf, MM, CC, 4 * CC);
    }

    ln_kernel<<<MM / 8, 256, 0, stream>>>(xbuf, lnf_s, lnf_b, hbuf);
    gemm_bf16<0><<<gl, 256, 0, stream>>>(hbuf, wteBF, nullptr, out, MM, VV, CC);

    loss_row_kernel<<<MM, 256, 0, stream>>>(out, tgt, rowloss);
    loss_mean_kernel<<<1, 256, 0, stream>>>(rowloss, out);
}
